// HyenaFilter2D_31988916420647
// MI455X (gfx1250) — compile-verified, matrix operand order checked by reading, not run on hardware
//
#include <hip/hip_runtime.h>
#include <hip/hip_bf16.h>
#include <math.h>

// ---------------------------------------------------------------------------
// HyenaFilter2D on gfx1250 (MI455X):
//   - implicit-filter MLP via v_wmma_f32_16x16x32_f16 (f16 in / f32 acc),
//     weights pre-swizzled into per-lane fragment order -> ds_load_b128
//   - Gaussian modulation fused in MLP epilogue
//   - 2D zero-padded FFT conv: Stockham radix-2 1024-pt FFTs in LDS,
//     source rows staged via global_load_async_to_lds_b128 (ASYNCcnt),
//     plane-by-plane so all spectra stay L2-resident (192 MB L2).
// ---------------------------------------------------------------------------

typedef __attribute__((ext_vector_type(16))) _Float16 v16h;
typedef __attribute__((ext_vector_type(8)))  float    v8f;

#define HW_H 512
#define HW_W 512
#define FN   1024            // padded FFT length (2*H)
#define PLANE (HW_H * HW_W)  // 262144

// Fragment table: 24 fragments x (32 lanes x 16 halves) = 12288 halves (24KB)
#define FRAG_L0  0           // 4 frags: nt 0..3            (W0p 32x64)
#define FRAG_L1  4           // 8 frags: kb{0,32} x nt 0..3 (W1 64x64)
#define FRAG_L2  12          // 8 frags: kb{0,32} x nt 0..3 (W2 64x64)
#define FRAG_OUT 20          // 4 frags: kb{0,32} x nt 0..1 (Wout 64x32)

// ============================ WMMA MLP =====================================

// Pre-swizzled fragment load: per-lane 32 contiguous bytes -> 2x ds_load_b128.
__device__ inline v16h load_frag(const _Float16* sw, int f, int lane) {
  const _Float16* p = sw + (f << 9) + (lane << 4);
  v16h r;
#pragma unroll
  for (int j = 0; j < 16; ++j) r[j] = p[j];
  return r;
}

// A fragment from a 16x64 f16 row-major LDS tile, K-offset kO (0 or 32).
__device__ inline v16h load_afrag(const _Float16* h, int kO, int lane) {
  const int m = lane & 15;
  const _Float16* p = h + m * 64 + kO + ((lane >> 4) << 3);
  v16h r;
#pragma unroll
  for (int j = 0; j < 8; ++j) {
    r[j]     = p[j];
    r[j + 8] = p[16 + j];
  }
  return r;
}

// D-layout (v8f x 4 ntiles) -> sin(freq*(x+b)) -> f16 row-major 16x64 in LDS.
__device__ inline void store_act(_Float16* h, const v8f* acc, const float* bias,
                                 const float* freq, int lane) {
  const int nlo = lane & 15;
  const int mo  = (lane >> 4) << 3;
#pragma unroll
  for (int nt = 0; nt < 4; ++nt) {
    int n = nt * 16 + nlo;
    float bb = bias[n], fq = freq[n];
#pragma unroll
    for (int v = 0; v < 8; ++v) {
      float x = acc[nt][v] + bb;
      h[(mo + v) * 64 + n] = (_Float16)__sinf(fq * x);
    }
  }
}

// Convert f32 weights -> f16, pre-swizzled into WMMA B-fragment layout:
// dst[f*512 + lane*16 + j] so each lane's 16 halves are contiguous (32B).
__global__ void convert_weights(const float* __restrict__ W0,
                                const float* __restrict__ W1,
                                const float* __restrict__ W2,
                                const float* __restrict__ Wout,
                                _Float16* __restrict__ dst) {
  int t = blockIdx.x * 256 + threadIdx.x;
  if (t >= 12288) return;
  int f    = t >> 9;          // fragment id
  int r    = t & 511;
  int lane = r >> 4;
  int j    = r & 15;
  int kin  = ((lane >> 4) << 3) + j + ((j >= 8) ? 8 : 0);  // K within 32-block
  int nlo  = lane & 15;
  float v;
  if (f < FRAG_L1) {                       // layer0: W0 10x64 zero-padded
    int K = kin, n = f * 16 + nlo;
    v = (K < 10) ? W0[K * 64 + n] : 0.0f;
  } else if (f < FRAG_L2) {                // layer1: W1 64x64
    int g = f - FRAG_L1;
    int K = ((g >= 4) ? 32 : 0) + kin, n = (g & 3) * 16 + nlo;
    v = W1[K * 64 + n];
  } else if (f < FRAG_OUT) {               // layer2: W2 64x64
    int g = f - FRAG_L2;
    int K = ((g >= 4) ? 32 : 0) + kin, n = (g & 3) * 16 + nlo;
    v = W2[K * 64 + n];
  } else {                                 // out: Wout 64x32
    int g = f - FRAG_OUT;
    int K = ((g >= 2) ? 32 : 0) + kin, n = (g & 1) * 16 + nlo;
    v = Wout[K * 32 + n];
  }
  dst[t] = (_Float16)v;
}

// One wave per 16-row tile of the [HW, *] MLP; 8 waves per block.
__global__ __launch_bounds__(256) void hyena_filter_mlp(
    const float* __restrict__ zh, const float* __restrict__ zw,
    const _Float16* __restrict__ wts,
    const float* __restrict__ b0, const float* __restrict__ b1,
    const float* __restrict__ b2, const float* __restrict__ freq,
    const float* __restrict__ deltas, float* __restrict__ kout) {
  __shared__ _Float16 sW[12288];
  __shared__ _Float16 sH[8][1024];   // per-wave 16x64 f16 transpose staging

  for (int i = threadIdx.x; i < 12288; i += 256) sW[i] = wts[i];
  __syncthreads();

  const int wave = threadIdx.x >> 5;
  const int lane = threadIdx.x & 31;
  const int tile = blockIdx.x * 8 + wave;
  const int rowBase = tile << 4;
  _Float16* hs = sH[wave];

  // ---- A0: padded positional embedding, K=10 valid of 32 ----
  v16h a0;
  {
    const int m = lane & 15;
    const int row = rowBase + m;
    const int iy = row >> 9, jx = row & 511;
    const int kbase = (lane >> 4) << 3;
#pragma unroll
    for (int j = 0; j < 16; ++j) {
      int K = kbase + j + ((j >= 8) ? 8 : 0);
      float v = 0.0f;
      if (K < 5)       v = zh[iy * 5 + K];
      else if (K < 10) v = zw[jx * 5 + (K - 5)];
      a0[j] = (_Float16)v;
    }
  }

  v8f acc[4];
  // ---- layer 0: [16x32] x [32x64] ----
#pragma unroll
  for (int nt = 0; nt < 4; ++nt) {
    v8f c = {};
    acc[nt] = __builtin_amdgcn_wmma_f32_16x16x32_f16(
        false, a0, false, load_frag(sW, FRAG_L0 + nt, lane), (short)0, c,
        false, false);
  }
  store_act(hs, acc, b0, freq, lane);
  asm volatile("s_wait_dscnt 0" ::: "memory");

  // ---- layer 1: [16x64] x [64x64] ----
  {
    v16h aA = load_afrag(hs, 0, lane);
    v16h aB = load_afrag(hs, 32, lane);
#pragma unroll
    for (int nt = 0; nt < 4; ++nt) {
      v8f c = {};
      c = __builtin_amdgcn_wmma_f32_16x16x32_f16(
          false, aA, false, load_frag(sW, FRAG_L1 + nt, lane), (short)0, c,
          false, false);
      acc[nt] = __builtin_amdgcn_wmma_f32_16x16x32_f16(
          false, aB, false, load_frag(sW, FRAG_L1 + 4 + nt, lane), (short)0, c,
          false, false);
    }
  }
  asm volatile("s_wait_dscnt 0" ::: "memory");
  store_act(hs, acc, b1, freq, lane);
  asm volatile("s_wait_dscnt 0" ::: "memory");

  // ---- layer 2: [16x64] x [64x64] ----
  {
    v16h aA = load_afrag(hs, 0, lane);
    v16h aB = load_afrag(hs, 32, lane);
#pragma unroll
    for (int nt = 0; nt < 4; ++nt) {
      v8f c = {};
      c = __builtin_amdgcn_wmma_f32_16x16x32_f16(
          false, aA, false, load_frag(sW, FRAG_L2 + nt, lane), (short)0, c,
          false, false);
      acc[nt] = __builtin_amdgcn_wmma_f32_16x16x32_f16(
          false, aB, false, load_frag(sW, FRAG_L2 + 4 + nt, lane), (short)0, c,
          false, false);
    }
  }
  asm volatile("s_wait_dscnt 0" ::: "memory");
  store_act(hs, acc, b2, freq, lane);
  asm volatile("s_wait_dscnt 0" ::: "memory");

  // ---- output layer: [16x64] x [64x32], no bias, no sin ----
  v8f oacc[2];
  {
    v16h aA = load_afrag(hs, 0, lane);
    v16h aB = load_afrag(hs, 32, lane);
#pragma unroll
    for (int nt = 0; nt < 2; ++nt) {
      v8f c = {};
      c = __builtin_amdgcn_wmma_f32_16x16x32_f16(
          false, aA, false, load_frag(sW, FRAG_OUT + nt, lane), (short)0, c,
          false, false);
      oacc[nt] = __builtin_amdgcn_wmma_f32_16x16x32_f16(
          false, aB, false, load_frag(sW, FRAG_OUT + 2 + nt, lane), (short)0, c,
          false, false);
    }
  }

  // ---- Gaussian modulation epilogue -> k[C,H,W] ----
  {
    const int nlo = lane & 15;
    const int mo  = (lane >> 4) << 3;
#pragma unroll
    for (int nt = 0; nt < 2; ++nt) {
      int cch = nt * 16 + nlo;
      float dl  = deltas[cch];
      float inv = 0.3989422804014327f / dl;   // 1/(delta*sqrt(2pi))
#pragma unroll
      for (int v = 0; v < 8; ++v) {
        int row = rowBase + mo + v;
        int iy = row >> 9, jx = row & 511;
        float xg = (float)iy - 256.0f;
        float yg = (float)jx - 256.0f;
        float t = (xg * xg + yg * yg) / dl;
        kout[(size_t)cch * PLANE + row] = oacc[nt][v] * (inv * __expf(-0.5f * t * t));
      }
    }
  }
}

// ============================ FFT pipeline =================================

__device__ inline float2 cmul(float2 a, float2 b) {
  return make_float2(a.x * b.x - a.y * b.y, a.x * b.y + a.y * b.x);
}

// Stockham radix-2 1024-pt FFT in LDS; 256 threads, 2 butterflies/thread/stage.
// Result ends in s0 (10 stages -> even number of ping-pongs).
__device__ inline void fft1024(float2* s0, float2* s1, int tid, float sign) {
  float2* src = s0;
  float2* dst = s1;
  int Ns = 1;
#pragma unroll
  for (int st = 0; st < 10; ++st) {
    __syncthreads();
    float invTwoNs = 1.0f / (float)(2 * Ns);
#pragma unroll
    for (int q = 0; q < 2; ++q) {
      int j = tid + q * 256;                    // 0..511
      float2 v0 = src[j];
      float2 v1 = src[j + 512];
      int r = j & (Ns - 1);
      float ang = sign * 6.283185307179586f * (float)r * invTwoNs;
      float sw, cw;
      __sincosf(ang, &sw, &cw);
      v1 = cmul(v1, make_float2(cw, sw));
      int idx = ((j >> (st)) << (st + 1)) + r;  // (j/Ns)*2Ns + (j%Ns)
      dst[idx]      = make_float2(v0.x + v1.x, v0.y + v1.y);
      dst[idx + Ns] = make_float2(v0.x - v1.x, v0.y - v1.y);
    }
    float2* tp = src; src = dst; dst = tp;
    Ns <<= 1;
  }
  __syncthreads();
}

// Zero rows 512..1023 of P0 once (invariant across all planes).
__global__ void zero_upper(float2* P0) {
  int i = blockIdx.x * 256 + threadIdx.x;       // covers 512*1024
  P0[(size_t)512 * FN + i] = make_float2(0.0f, 0.0f);
}

// Forward FFT of one zero-padded real row (rows 0..511 of the padded grid).
// The 2KB source row is DMA'd into LDS via global_load_async_to_lds_b128.
__global__ __launch_bounds__(256) void rowfft_real(const float* __restrict__ src,
                                                   float2* __restrict__ dst) {
  __shared__ float2 s0[FN], s1[FN];
  __shared__ float srow[HW_W];
  const int r = blockIdx.x;
  const int tid = threadIdx.x;
  // Async DMA: 128 lanes x 16B = 2KB row, ASYNCcnt-tracked, no VGPR staging.
  if (tid < 128) {
    const float* gp = src + (size_t)r * HW_W + tid * 4;
    unsigned ldsOff = (unsigned)(size_t)&srow[tid * 4];
    asm volatile("global_load_async_to_lds_b128 %0, %1, off"
                 :: "v"(ldsOff), "v"(gp) : "memory");
  }
  asm volatile("s_wait_asynccnt 0x0" ::: "memory");
  __syncthreads();
#pragma unroll
  for (int q = 0; q < 4; ++q) {
    int i = tid + q * 256;
    float v = (i < HW_W) ? srow[i & (HW_W - 1)] : 0.0f;
    s0[i] = make_float2(v, 0.0f);
  }
  fft1024(s0, s1, tid, -1.0f);
#pragma unroll
  for (int q = 0; q < 4; ++q) {
    int i = tid + q * 256;
    dst[(size_t)r * FN + i] = s0[i];
  }
}

// Forward column FFT with output scaling (builds k_f = fft2(k)/(FH*FW)).
__global__ __launch_bounds__(256) void colfft_scale(const float2* __restrict__ src,
                                                    float2* __restrict__ dst,
                                                    float scale) {
  __shared__ float2 s0[FN], s1[FN];
  const int col = blockIdx.x;
  const int tid = threadIdx.x;
#pragma unroll
  for (int q = 0; q < 4; ++q) {
    int i = tid + q * 256;
    s0[i] = src[(size_t)i * FN + col];
  }
  fft1024(s0, s1, tid, -1.0f);
#pragma unroll
  for (int q = 0; q < 4; ++q) {
    int i = tid + q * 256;
    float2 v = s0[i];
    dst[(size_t)i * FN + col] = make_float2(v.x * scale, v.y * scale);
  }
}

// Fused: forward col FFT -> multiply by k_f column -> inverse col FFT.
__global__ __launch_bounds__(256) void col_conv(const float2* __restrict__ P0,
                                                const float2* __restrict__ KF,
                                                float2* __restrict__ P1) {
  __shared__ float2 s0[FN], s1[FN];
  const int col = blockIdx.x;
  const int tid = threadIdx.x;
#pragma unroll
  for (int q = 0; q < 4; ++q) {
    int i = tid + q * 256;
    s0[i] = P0[(size_t)i * FN + col];
  }
  fft1024(s0, s1, tid, -1.0f);
#pragma unroll
  for (int q = 0; q < 4; ++q) {             // each thread touches its own i's
    int i = tid + q * 256;
    s0[i] = cmul(s0[i], KF[(size_t)i * FN + col]);
  }
  fft1024(s0, s1, tid, 1.0f);               // leading barrier orders the mul
  const float invN = 1.0f / (float)FN;
#pragma unroll
  for (int q = 0; q < 4; ++q) {
    int i = tid + q * 256;
    float2 v = s0[i];
    P1[(size_t)i * FN + col] = make_float2(v.x * invN, v.y * invN);
  }
}

// Inverse row FFT, crop to [0,512), add u*D bias path, write output plane.
__global__ __launch_bounds__(256) void irow_out(const float2* __restrict__ P1,
                                                const float* __restrict__ uplane,
                                                const float* __restrict__ Dv,
                                                int c, float* __restrict__ outp) {
  __shared__ float2 s0[FN], s1[FN];
  const int r = blockIdx.x;                 // 0..511
  const int tid = threadIdx.x;
#pragma unroll
  for (int q = 0; q < 4; ++q) {
    int i = tid + q * 256;
    s0[i] = P1[(size_t)r * FN + i];
  }
  fft1024(s0, s1, tid, 1.0f);
  const float invN = 1.0f / (float)FN;
  const float dc = Dv[c];
#pragma unroll
  for (int q = 0; q < 2; ++q) {             // only cols < 512 survive the crop
    int i = tid + q * 256;
    size_t o = (size_t)r * HW_W + i;
    outp[o] = s0[i].x * invN + uplane[o] * dc;
  }
}

// ============================ launch =======================================

extern "C" void kernel_launch(void* const* d_in, const int* in_sizes, int n_in,
                              void* d_out, int out_size, void* d_ws,
                              size_t ws_size, hipStream_t stream) {
  const float* u    = (const float*)d_in[0];
  const float* zh   = (const float*)d_in[1];
  const float* zw   = (const float*)d_in[2];
  const float* W0   = (const float*)d_in[3];
  const float* b0   = (const float*)d_in[4];
  const float* W1   = (const float*)d_in[5];
  const float* b1   = (const float*)d_in[6];
  const float* W2   = (const float*)d_in[7];
  const float* b2   = (const float*)d_in[8];
  const float* Wout = (const float*)d_in[9];
  const float* freq = (const float*)d_in[10];
  const float* dts  = (const float*)d_in[11];
  const float* Dv   = (const float*)d_in[12];
  float* out = (float*)d_out;

  char* ws = (char*)d_ws;
  _Float16* wts = (_Float16*)ws;                             // 24 KB (pad 32 KB)
  float*  kbuf  = (float*)(ws + 32768);                      // 32 MB  k[C,H,W]
  float2* P0    = (float2*)(ws + 32768 + (size_t)32 * PLANE * 4);
  float2* P1    = P0 + (size_t)FN * FN;                      // 8 MB each
  float2* KF    = P1 + (size_t)FN * FN;

  // 1) weights -> f16, pre-swizzled into WMMA fragment order
  convert_weights<<<48, 256, 0, stream>>>(W0, W1, W2, Wout, wts);

  // 2) implicit filter MLP (WMMA) + Gaussian modulation -> kbuf
  hyena_filter_mlp<<<2048, 256, 0, stream>>>(zh, zw, wts, b0, b1, b2, freq,
                                             dts, kbuf);

  // 3) zero-padded upper half of P0 (invariant across all planes)
  zero_upper<<<2048, 256, 0, stream>>>(P0);

  // 4) per-channel spectrum, then per-(b,c) FFT conv. Everything between the
  //    rowfft and irow stages lives in P0/P1/KF (24 MB) -> L2 resident.
  const float kfScale = 1.0f / ((float)FN * (float)FN);
  for (int c = 0; c < 32; ++c) {
    rowfft_real<<<512, 256, 0, stream>>>(kbuf + (size_t)c * PLANE, P0);
    colfft_scale<<<1024, 256, 0, stream>>>(P0, KF, kfScale);
    for (int b = 0; b < 2; ++b) {
      const float* uplane = u + ((size_t)b * 32 + c) * PLANE;
      float* outplane = out + ((size_t)b * 32 + c) * PLANE;
      rowfft_real<<<512, 256, 0, stream>>>(uplane, P0);
      col_conv<<<1024, 256, 0, stream>>>(P0, KF, P1);
      irow_out<<<512, 256, 0, stream>>>(P1, uplane, Dv, c, outplane);
    }
  }
}
